// patchEmbedding_17205638988437
// MI455X (gfx1250) — matrix operand order repaired and checked by reading, not compiled, not run on hardware
//
#include <hip/hip_runtime.h>
#include <hip/hip_bf16.h>

// ---------- problem constants (from reference) ----------
#define BATCH 32
#define TSEQ  4096
#define HDIM  512
#define WDIM  512
#define PSZ   16
#define PADP  12            // P/2 + P/4
#define KTOT  256           // P*P*CH
#define EMB   768
#define MTOT  (BATCH * TSEQ)      // 131072 rows
#define MBLK  64                  // patches per workgroup
#define NT_ALL   48               // 768 / 16 N-tiles
#define KSTEPS   8                // 256 / 32
#define ASTRIDE  264              // 256 + 8 bf16 pad (keeps 16B align, kills bank conflicts)
#define NT_WAVE  24               // N-tiles per wave (48 / 2 N-waves)

typedef __attribute__((ext_vector_type(16))) __bf16 v16bf;
typedef __attribute__((ext_vector_type(8)))  __bf16 v8bf;
typedef __attribute__((ext_vector_type(8)))  float  v8f;

// ---------------------------------------------------------------------------
// Kernel 1: repack Wlin (f32 [EMB][KTOT], row-major) into bf16 fragments.
// Layout: Wp[((ntile*8 + ks)*32 + lane)*16 + e]
//   lane<16 : column N = ntile*16 + lane,    K = ks*32 + 0..15  (e)
//   lane>=16: column N = ntile*16 + lane-16, K = ks*32 + 16..31 (e)
// K is contiguous per lane half, the same order the A fragment uses, so a wave's B
// fragment is one contiguous, coalesced 1KB load.
// ---------------------------------------------------------------------------
__global__ void pack_weights_bf16(const float* __restrict__ Wlin,
                                  __bf16* __restrict__ Wp) {
    int idx   = blockIdx.x * 256 + threadIdx.x;     // 0 .. 196607
    int e     = idx & 15;
    int lane  = (idx >> 4) & 31;
    int ks    = (idx >> 9) & 7;
    int ntile = idx >> 12;                          // 0 .. 47
    int n16   = lane & 15;
    int khalf = (lane < 16) ? 0 : 16;
    int Ncol  = ntile * 16 + n16;
    int k     = ks * 32 + khalf + e;
    Wp[idx] = (__bf16)Wlin[Ncol * KTOT + k];
}

// ---------------------------------------------------------------------------
// Kernel 2: gather 64 patches -> LDS(bf16), then WMMA GEMM tile 64x768.
// 8 waves arranged 4(M) x 2(N); each wave: 16 rows x 384 cols.
// ---------------------------------------------------------------------------
__global__ void __launch_bounds__(256)
patch_embed_wmma(const float* __restrict__ input,     // [B][1][512][512]
                 const int*   __restrict__ kp,        // [B][T][2] (x,y)
                 const int*   __restrict__ sh,        // [B][T][2]
                 const __bf16* __restrict__ Wp,       // packed bf16 weights
                 const float* __restrict__ blin,      // [EMB]
                 float*       __restrict__ out)       // [B][T][1][EMB]
{
    __shared__ __bf16 Alds[MBLK * ASTRIDE];

    const int tid  = threadIdx.x;
    const int wgM  = blockIdx.x * MBLK;

    // ---------------- phase 1: gather patches into LDS as bf16 -------------
    {
        const int j  = tid & 15;        // column within patch (coalesced)
        const int pg = tid >> 4;        // 16 thread-groups
        #pragma unroll
        for (int pp = 0; pp < 4; ++pp) {
            const int p = pg * 4 + pp;              // patch 0..63 in block
            const int m = wgM + p;                  // global row
            const int b = m >> 12;                  // m / TSEQ
            const int x0 = kp[2 * m + 0] + sh[2 * m + 0];
            const int y0 = kp[2 * m + 1] + sh[2 * m + 1];
            const float* img = input + (size_t)b * (HDIM * WDIM);
            const int gx = x0 + j - PADP;
            const bool xok = (gx >= 0) && (gx < WDIM);
            #pragma unroll
            for (int i = 0; i < PSZ; ++i) {
                const int gy = y0 + i - PADP;
                float v = 0.0f;
                if (xok && gy >= 0 && gy < HDIM) v = img[gy * WDIM + gx];
                Alds[p * ASTRIDE + i * PSZ + j] = (__bf16)v;
            }
        }
    }
    __syncthreads();

    // ---------------- phase 2: WMMA GEMM -----------------------------------
    const int lane   = tid & 31;
    const int waveId = tid >> 5;
    const int m_wave = waveId & 3;          // 0..3 -> 16-row tile
    const int n_wave = waveId >> 2;         // 0..1 -> 384-col half
    const int n16    = lane & 15;
    const int arow   = m_wave * 16 + n16;               // A row in LDS
    const int klo0   = (lane < 16) ? 0 : 16;            // A frag K offset
    const int ntg0   = n_wave * NT_WAVE;                // first global N-tile

    v8f acc[NT_WAVE];
    #pragma unroll
    for (int nt = 0; nt < NT_WAVE; ++nt) acc[nt] = (v8f){};

    // bias per lane per tile (columns: lanes 0-15 == lanes 16-31)
    float bv[NT_WAVE];
    #pragma unroll
    for (int nt = 0; nt < NT_WAVE; ++nt)
        bv[nt] = blin[(ntg0 + nt) * 16 + n16];

    for (int ks = 0; ks < KSTEPS; ++ks) {
        // A fragment: 16x32, K contiguous per lane half (same order as packed B)
        const __bf16* aptr = &Alds[arow * ASTRIDE + ks * 32 + klo0];
        v8bf alo = *(const v8bf*)aptr;           // K klo0 .. klo0+7
        v8bf ahi = *(const v8bf*)(aptr + 8);     // K klo0+8 .. klo0+15
        v16bf a;
        #pragma unroll
        for (int i = 0; i < 8; ++i) { a[i] = alo[i]; a[8 + i] = ahi[i]; }

        #pragma unroll
        for (int nt = 0; nt < NT_WAVE; ++nt) {
            const int ntg = ntg0 + nt;
            const v16bf bfrg =
                *(const v16bf*)(Wp + ((size_t)(ntg * KSTEPS + ks) * 32 + lane) * 16);
            acc[nt] = __builtin_amdgcn_wmma_f32_16x16x32_bf16(
                /*neg_a=*/false, a, /*neg_b=*/false, bfrg,
                /*c_mod=*/(short)0, acc[nt],
                /*reuse_a=*/false, /*reuse_b=*/false);
        }
    }

    // ---------------- phase 3: bias + store --------------------------------
    // C/D layout: VGPR r: lanes 0-15 -> M=r, lanes 16-31 -> M=8+r; N=lane&15
    const int moff    = (lane < 16) ? 0 : 8;
    const int rowBase = wgM + m_wave * 16 + moff;
    #pragma unroll
    for (int nt = 0; nt < NT_WAVE; ++nt) {
        const int col = (ntg0 + nt) * 16 + n16;
        #pragma unroll
        for (int r = 0; r < 8; ++r) {
            out[(size_t)(rowBase + r) * EMB + col] = acc[nt][r] + bv[nt];
        }
    }
}

extern "C" void kernel_launch(void* const* d_in, const int* in_sizes, int n_in,
                              void* d_out, int out_size, void* d_ws, size_t ws_size,
                              hipStream_t stream) {
    (void)in_sizes; (void)n_in; (void)out_size; (void)ws_size;
    const float* input = (const float*)d_in[0];
    const int*   kp    = (const int*)  d_in[1];
    const int*   sh    = (const int*)  d_in[2];
    const float* Wlin  = (const float*)d_in[3];
    const float* blin  = (const float*)d_in[4];
    float*  out = (float*)d_out;
    __bf16* Wp  = (__bf16*)d_ws;    // 48*8*32*16 bf16 = 384 KB scratch

    // 1) repack weights to bf16 fragment layout (196608 elements)
    pack_weights_bf16<<<dim3(NT_ALL * KSTEPS * 32 * 16 / 256), dim3(256), 0, stream>>>(Wlin, Wp);

    // 2) gather + GEMM: 131072/64 = 2048 workgroups
    patch_embed_wmma<<<dim3(MTOT / MBLK), dim3(256), 0, stream>>>(
        input, kp, sh, Wp, blin, out);
}
